// SaccadicAttentionLayer_45518063403229
// MI455X (gfx1250) — matrix operand order repaired and checked by reading, not compiled, not run on hardware
//
#include <hip/hip_runtime.h>
#include <math.h>

// ---------------- types for WMMA ----------------
typedef __attribute__((ext_vector_type(16))) __bf16 v16bf;
typedef __attribute__((ext_vector_type(8)))  __bf16 v8bf;
typedef __attribute__((ext_vector_type(8)))  float  v8f;

#define CAT16(lo, hi) __builtin_shufflevector(lo, hi, 0,1,2,3,4,5,6,7,8,9,10,11,12,13,14,15)

// Model constants
#define BB   4
#define SEQ  4096
#define DD   1024
#define NHD  16
#define DH   64
#define MBLK 512          // SEQ / 8
#define NSAC 3

__device__ __forceinline__ float geluf(float x) {
    return 0.5f * x * (1.0f + erff(x * 0.70710678118654752f));
}

// pack 16 fp32 (from global) into 16 bf16 in LDS as two 16B stores
__device__ __forceinline__ void fill16(const float* __restrict__ g, __bf16* dst) {
    float4 f0 = *(const float4*)(g);
    float4 f1 = *(const float4*)(g + 4);
    float4 f2 = *(const float4*)(g + 8);
    float4 f3 = *(const float4*)(g + 12);
    v8bf p0, p1;
    p0[0] = (__bf16)f0.x; p0[1] = (__bf16)f0.y; p0[2] = (__bf16)f0.z; p0[3] = (__bf16)f0.w;
    p0[4] = (__bf16)f1.x; p0[5] = (__bf16)f1.y; p0[6] = (__bf16)f1.z; p0[7] = (__bf16)f1.w;
    p1[0] = (__bf16)f2.x; p1[1] = (__bf16)f2.y; p1[2] = (__bf16)f2.z; p1[3] = (__bf16)f2.w;
    p1[4] = (__bf16)f3.x; p1[5] = (__bf16)f3.y; p1[6] = (__bf16)f3.z; p1[7] = (__bf16)f3.w;
    *(v8bf*)(dst)     = p0;
    *(v8bf*)(dst + 8) = p1;
}

// ==================================================================
// WMMA GEMM: C[M,N] = A[M,K] @ W[N,K]^T + bias[N]  (+gelu / +R residual)
// LDS-staged, double-buffered. Block = 128 threads (4 waves) computing a
// 64(M) x 64(N) tile; each wave owns a 16x64 strip. Per k-step of 32 the
// block cooperatively stages A(64x32) and W(64x32) as bf16 in LDS (16KB
// total for 2 buffers). All 10 fragment ds_loads are issued before the
// 4 WMMAs so they pipeline under partial DS waits. No masking in the hot
// loop: OOB rows/cols are index-clamped (safe addresses) and skipped only
// at store time, since C[m,n] depends solely on A row m and W col n.
// EXEC is all-ones at every WMMA.
// mode: 0 = bias, 1 = bias+gelu, 2 = bias + R[m,n]
// K must be a multiple of 32 (all uses: 1024/2048/3072/4096).
// ==================================================================
__global__ __launch_bounds__(128)
void gemm_wmma_kernel(const float* __restrict__ A, const float* __restrict__ W,
                      const float* __restrict__ bias, const float* __restrict__ R,
                      float* __restrict__ C, int M, int N, int K, int mode)
{
    __shared__ __bf16 Asm[2][64 * 32];
    __shared__ __bf16 Wsm[2][64 * 32];

    const int tid   = threadIdx.x;
    const int lane  = tid & 31;
    const int wave  = tid >> 5;
    const int tileM = blockIdx.y << 6;
    const int tileN = blockIdx.x << 6;
    const int mrow  = lane & 15;
    const int half  = lane >> 4;          // 0 or 1

    // ---- cooperative fill mapping: thread -> (row 0..63, k-half 0/16)
    const int fr = tid >> 1;
    const int fk = (tid & 1) << 4;
    int ar = tileM + fr; if (ar > M - 1) ar = M - 1;
    int wc = tileN + fr; if (wc > N - 1) wc = N - 1;
    const float* Ag = A + (size_t)ar * K + fk;
    const float* Wg = W + (size_t)wc * K + fk;
    const int fo = fr * 32 + fk;

    // prologue: stage k-chunk 0 into buffer 0
    fill16(Ag, &Asm[0][fo]);
    fill16(Wg, &Wsm[0][fo]);
    __syncthreads();

    v8f acc[4];
    acc[0] = v8f{}; acc[1] = v8f{}; acc[2] = v8f{}; acc[3] = v8f{};

    int buf = 0;
    for (int k0 = 0; k0 < K; k0 += 32) {
        const int nbuf = buf ^ 1;
        if (k0 + 32 < K) {                 // stage next chunk while computing
            fill16(Ag + k0 + 32, &Asm[nbuf][fo]);
            fill16(Wg + k0 + 32, &Wsm[nbuf][fo]);
        }
        // ---- issue ALL fragment loads first so DS waits pipeline ----
        // A fragment (16x32 bf16), ISA 7.12.2 layout:
        // lanes 0-15: K 0..7 & 16..23 ; lanes 16-31: K 8..15 & 24..31
        const __bf16* Arow = &Asm[buf][(wave * 16 + mrow) * 32 + half * 8];
        v8bf alo = *(const v8bf*)(Arow);
        v8bf ahi = *(const v8bf*)(Arow + 16);
        // 4 B fragments (32x16 bf16): lanes 0-15 K 0..7 & 16..23 ; lanes 16-31 K 8..15 & 24..31
        v8bf blo[4], bhi[4];
#pragma unroll
        for (int j = 0; j < 4; ++j) {
            const __bf16* Wrow = &Wsm[buf][(j * 16 + mrow) * 32 + half * 8];
            blo[j] = *(const v8bf*)(Wrow);
            bhi[j] = *(const v8bf*)(Wrow + 16);
        }
        v16bf af = CAT16(alo, ahi);
#pragma unroll
        for (int j = 0; j < 4; ++j) {
            v16bf bfv = CAT16(blo[j], bhi[j]);
            acc[j] = __builtin_amdgcn_wmma_f32_16x16x32_bf16(
                false, af, false, bfv, (short)0, acc[j], false, false);
        }
        __syncthreads();                   // fill(nbuf) done; reads of buf done
        buf = nbuf;
    }

    // ---- epilogue: C row = tileM + wave*16 + v + 8*half, col = tileN + j*16 + mrow
#pragma unroll
    for (int j = 0; j < 4; ++j) {
        const int col = tileN + j * 16 + mrow;
        if (col >= N) continue;
        const float bv = bias ? bias[col] : 0.0f;
#pragma unroll
        for (int v = 0; v < 8; ++v) {
            const int row = tileM + wave * 16 + half * 8 + v;
            if (row >= M) continue;
            float val = acc[j][v] + bv;
            if (mode == 1)      val = geluf(val);
            else if (mode == 2) val += R[(size_t)row * N + col];
            C[(size_t)row * N + col] = val;
        }
    }
}

// ==================================================================
// LayerNorm over D=1024 (per row), optional additive "extra" row
// (used for pos-embedding add): y += extra[(row % extra_mod)*D + d]
// ==================================================================
__global__ void ln_kernel(const float* __restrict__ X, const float* __restrict__ g,
                          const float* __restrict__ be, const float* __restrict__ extra,
                          int extra_mod, float* __restrict__ Y)
{
    const int row = blockIdx.x;
    const int tid = threadIdx.x;
    const float* xr = X + (size_t)row * DD;
    __shared__ float r1[256], r2[256];
    float s = 0.f, sq = 0.f;
    for (int d = tid; d < DD; d += 256) { float v = xr[d]; s += v; sq += v * v; }
    r1[tid] = s; r2[tid] = sq; __syncthreads();
    for (int st = 128; st > 0; st >>= 1) {
        if (tid < st) { r1[tid] += r1[tid + st]; r2[tid] += r2[tid + st]; }
        __syncthreads();
    }
    const float mean = r1[0] * (1.0f / DD);
    const float var  = r2[0] * (1.0f / DD) - mean * mean;
    const float rstd = rsqrtf(var + 1e-5f);
    float* yr = Y + (size_t)row * DD;
    const float* er = extra ? (extra + (size_t)(row % extra_mod) * DD) : nullptr;
    for (int d = tid; d < DD; d += 256) {
        float v = (xr[d] - mean) * rstd * g[d] + be[d];
        if (er) v += er[d];
        yr[d] = v;
    }
}

// ---- peripheral score matvec: s[row] = dot(h[row,:], wp_w) + wp_b ----
__global__ __launch_bounds__(32)
void matvec_kernel(const float* __restrict__ X, const float* __restrict__ w,
                   const float* __restrict__ b0, float* __restrict__ out)
{
    const int row = blockIdx.x;
    const int lane = threadIdx.x;
    const float* xr = X + (size_t)row * DD;
    float s = 0.f;
    for (int e = lane; e < DD; e += 32) s += xr[e] * w[e];
    for (int off = 16; off > 0; off >>= 1) s += __shfl_down(s, off, 32);
    if (lane == 0) out[row] = s + b0[0];
}

// ---- peripheral block stats: softmax weights, wmean/wstd/bmax -> cat[3D] ----
__global__ void periph_stats_kernel(const float* __restrict__ h, const float* __restrict__ s,
                                    float* __restrict__ cat)
{
    const int bm = blockIdx.x;            // b*512 + blk
    const int b = bm >> 9, blk = bm & 511;
    const int tid = threadIdx.x;
    __shared__ float w[8];
    if (tid == 0) {
        float sv[8], mx = -3.4e38f;
        for (int j = 0; j < 8; ++j) { sv[j] = s[b * SEQ + blk * 8 + j]; mx = fmaxf(mx, sv[j]); }
        float sum = 0.f;
        for (int j = 0; j < 8; ++j) { sv[j] = expf(sv[j] - mx); sum += sv[j]; }
        for (int j = 0; j < 8; ++j) w[j] = sv[j] / sum;
    }
    __syncthreads();
    const float* base = h + ((size_t)b * SEQ + blk * 8) * DD;
    float* cr = cat + (size_t)bm * (3 * DD);
    for (int d = tid; d < DD; d += 256) {
        float xs[8], xm = 0.f, bmax = -3.4e38f;
        for (int j = 0; j < 8; ++j) {
            float v = base[(size_t)j * DD + d];
            xs[j] = v; xm += w[j] * v; bmax = fmaxf(bmax, v);
        }
        float xv = 0.f;
        for (int j = 0; j < 8; ++j) { float dd = xs[j] - xm; xv += w[j] * dd * dd; }
        cr[d]          = xm;
        cr[DD + d]     = sqrtf(xv + 1e-8f);
        cr[2 * DD + d] = bmax;
    }
}

// ---- state = mean over 512 pmap rows ----
__global__ void mean_kernel(const float* __restrict__ pmap, float* __restrict__ state)
{
    const int idx = blockIdx.x * 256 + threadIdx.x;
    if (idx >= BB * DD) return;
    const int b = idx >> 10, d = idx & 1023;
    const float* p = pmap + (size_t)b * MBLK * DD + d;
    float s = 0.f;
    for (int m = 0; m < MBLK; ++m) s += p[(size_t)m * DD];
    state[idx] = s * (1.0f / MBLK);
}

// ---- ctrl scores: out[b*512+m] = dot(q[b,:], k[b*512+m,:]) * scale ----
__global__ __launch_bounds__(32)
void dot_row_kernel(const float* __restrict__ q, const float* __restrict__ k,
                    float* __restrict__ out, float scale)
{
    const int row = blockIdx.x;           // b*512 + m
    const int b = row >> 9;
    const int lane = threadIdx.x;
    const float* qr = q + (size_t)b * DD;
    const float* kr = k + (size_t)row * DD;
    float s = 0.f;
    for (int e = lane; e < DD; e += 32) s += qr[e] * kr[e];
    for (int off = 16; off > 0; off >>= 1) s += __shfl_down(s, off, 32);
    if (lane == 0) out[row] = s * scale;
}

// ---- argmax over 512 scores -> clipped window start ----
__global__ void argmax_kernel(const float* __restrict__ scores, int* __restrict__ starts)
{
    const int b = blockIdx.x;
    const int tid = threadIdx.x;
    __shared__ float bv[256];
    __shared__ int   bi[256];
    float best = -3.4e38f; int bidx = 0;
    for (int m = tid; m < MBLK; m += 256) {
        float v = scores[b * MBLK + m];
        if (v > best) { best = v; bidx = m; }
    }
    bv[tid] = best; bi[tid] = bidx; __syncthreads();
    for (int s2 = 128; s2 > 0; s2 >>= 1) {
        if (tid < s2) {
            if (bv[tid + s2] > bv[tid] ||
                (bv[tid + s2] == bv[tid] && bi[tid + s2] < bi[tid])) {
                bv[tid] = bv[tid + s2]; bi[tid] = bi[tid + s2];
            }
        }
        __syncthreads();
    }
    if (tid == 0) {
        int fp = bi[0] * 8;
        int st = fp - 32;
        if (st < 0) st = 0;
        if (st > SEQ - 64) st = SEQ - 64;
        starts[b] = st;
    }
}

// ---- gather 64-row window from h into acc slot t ----
__global__ void gather_window_kernel(const float* __restrict__ h, const int* __restrict__ starts,
                                     float* __restrict__ acc, int t)
{
    const int bw = blockIdx.x;            // b*64 + w
    const int b = bw >> 6, wrow = bw & 63;
    const int s = starts[b];
    const float* src = h + ((size_t)b * SEQ + s + wrow) * DD;
    float* dst = acc + ((size_t)(b * 192 + t * 64 + wrow)) * DD;
    for (int d = threadIdx.x; d < DD; d += 256) dst[d] = src[d];
}

// ---- ctx = [state_row, acc rows 0..Lc-2] per batch ----
__global__ void build_ctx_kernel(const float* __restrict__ state, const float* __restrict__ acc,
                                 float* __restrict__ ctx, int Lc)
{
    const int idx = blockIdx.x * 256 + threadIdx.x;
    const int total = BB * Lc * DD;
    if (idx >= total) return;
    const int d = idx & 1023;
    const int l = (idx >> 10) % Lc;
    const int b = idx / (Lc * DD);
    ctx[idx] = (l == 0) ? state[b * DD + d]
                        : acc[((size_t)b * 192 + (l - 1)) * DD + d];
}

__global__ void cls_copy_kernel(const float* __restrict__ ctx, float* __restrict__ cls, int Lc)
{
    const int idx = blockIdx.x * 256 + threadIdx.x;
    if (idx >= BB * DD) return;
    const int b = idx >> 10;
    cls[idx] = ctx[(size_t)b * Lc * DD + (idx & 1023)];
}

// ==================================================================
// Attention core (dh=64, 16 heads): scores, softmax, weighted V.
// Row addressing: q row = b*lqs + i (stride qstride), k/v row = b*lks + j.
// O is contiguous (B, Lq, 1024) merged-head layout.
// ==================================================================
__global__ __launch_bounds__(64)
void attn_kernel(const float* __restrict__ Q, int qstride, int lqs,
                 const float* __restrict__ Kp, const float* __restrict__ Vp,
                 int kvstride, int lks,
                 float* __restrict__ O, int Lq, int Lk, float scale)
{
    __shared__ float qs[DH];
    __shared__ float sc[224];             // Lk <= 193
    const int idx = blockIdx.x;
    const int i = idx % Lq;
    const int h = (idx / Lq) % NHD;
    const int b = idx / (Lq * NHD);
    const int tid = threadIdx.x;

    qs[tid] = Q[(size_t)(b * lqs + i) * qstride + h * DH + tid];
    __syncthreads();

    for (int j = tid; j < Lk; j += 64) {
        const float* kr = Kp + (size_t)(b * lks + j) * kvstride + h * DH;
        float d = 0.f;
        for (int e = 0; e < DH; ++e) d += qs[e] * kr[e];
        sc[j] = d * scale;
    }
    __syncthreads();
    float mx = -3.4e38f;
    for (int j = 0; j < Lk; ++j) mx = fmaxf(mx, sc[j]);
    __syncthreads();
    for (int j = tid; j < Lk; j += 64) sc[j] = expf(sc[j] - mx);
    __syncthreads();
    float sum = 0.f;
    for (int j = 0; j < Lk; ++j) sum += sc[j];
    const float inv = 1.0f / sum;
    float o = 0.f;
    for (int j = 0; j < Lk; ++j)
        o += sc[j] * Vp[(size_t)(b * lks + j) * kvstride + h * DH + tid];
    O[(size_t)(b * Lq + i) * DD + h * DH + tid] = o * inv;
}

// ---- gate: alpha = sigmoid( dot(gelu(step*g1w + g1b), g2w) + g2b ) ----
__global__ void alpha_kernel(const float* __restrict__ g1w, const float* __restrict__ g1b,
                             const float* __restrict__ g2w, const float* __restrict__ g2b,
                             float step, float* __restrict__ alpha)
{
    __shared__ float red[256];
    const int tid = threadIdx.x;
    float p = 0.f;
    for (int d = tid; d < DD; d += 256) {
        float hg = geluf(step * g1w[d] + g1b[d]);
        p += hg * g2w[d];
    }
    red[tid] = p; __syncthreads();
    for (int s = 128; s > 0; s >>= 1) {
        if (tid < s) red[tid] += red[tid + s];
        __syncthreads();
    }
    if (tid == 0) alpha[0] = 1.0f / (1.0f + expf(-(red[0] + g2b[0])));
}

// ---- pmap += alpha * delta ----
__global__ void axpy_kernel(const float* __restrict__ alpha, const float* __restrict__ dlt,
                            float* __restrict__ pm, int total)
{
    const int idx = blockIdx.x * 256 + threadIdx.x;
    if (idx >= total) return;
    pm[idx] += alpha[0] * dlt[idx];
}

// ---- out[b,n,:] = x[b,n,:] + orow[b,:] ----
__global__ void broadcast_add_kernel(const float* __restrict__ x, const float* __restrict__ orow,
                                     float* __restrict__ out)
{
    const size_t idx = (size_t)blockIdx.x * 256 + threadIdx.x;
    if (idx >= (size_t)BB * SEQ * DD) return;
    const int d = (int)(idx & 1023);
    const int b = (int)(idx / ((size_t)SEQ * DD));
    out[idx] = x[idx] + orow[b * DD + d];
}

// ==================================================================
extern "C" void kernel_launch(void* const* d_in, const int* in_sizes, int n_in,
                              void* d_out, int out_size, void* d_ws, size_t ws_size,
                              hipStream_t stream)
{
    (void)in_sizes; (void)n_in; (void)out_size; (void)ws_size;
    // ---- inputs (setup_inputs dict order, nested dicts flattened in order) ----
    const float* x        = (const float*)d_in[0];
    const float* wp_w     = (const float*)d_in[1];
    const float* wp_b     = (const float*)d_in[2];
    const float* sp_w     = (const float*)d_in[3];
    const float* sp_b     = (const float*)d_in[4];
    const float* p_ln_g   = (const float*)d_in[5];
    const float* p_ln_b   = (const float*)d_in[6];
    const float* pos      = (const float*)d_in[7];
    const float* cq_w     = (const float*)d_in[8];
    const float* cq_b     = (const float*)d_in[9];
    const float* ck_w     = (const float*)d_in[10];
    const float* ck_b     = (const float*)d_in[11];
    const float* fa_in_w  = (const float*)d_in[12];
    const float* fa_in_b  = (const float*)d_in[13];
    const float* fa_out_w = (const float*)d_in[14];
    const float* fa_out_b = (const float*)d_in[15];
    const float* f_ln1_g  = (const float*)d_in[16];
    const float* f_ln1_b  = (const float*)d_in[17];
    const float* f_ln2_g  = (const float*)d_in[18];
    const float* f_ln2_b  = (const float*)d_in[19];
    const float* f1_w     = (const float*)d_in[20];
    const float* f1_b     = (const float*)d_in[21];
    const float* f2_w     = (const float*)d_in[22];
    const float* f2_b     = (const float*)d_in[23];
    const float* ma_in_w  = (const float*)d_in[24];
    const float* ma_in_b  = (const float*)d_in[25];
    const float* ma_out_w = (const float*)d_in[26];
    const float* ma_out_b = (const float*)d_in[27];
    const float* mn_g     = (const float*)d_in[28];
    const float* mn_b     = (const float*)d_in[29];
    const float* g1w      = (const float*)d_in[30];
    const float* g1b      = (const float*)d_in[31];
    const float* g2w      = (const float*)d_in[32];
    const float* g2b      = (const float*)d_in[33];
    const float* out_w    = (const float*)d_in[34];
    const float* out_b    = (const float*)d_in[35];
    const float* outn_g   = (const float*)d_in[36];
    const float* outn_b   = (const float*)d_in[37];
    const float* ln1_g    = (const float*)d_in[38];
    const float* ln1_b    = (const float*)d_in[39];
    const float* ln2_g    = (const float*)d_in[40];
    const float* ln2_b    = (const float*)d_in[41];
    const float* m1_w     = (const float*)d_in[42];
    const float* m1_b     = (const float*)d_in[43];
    const float* m2_w     = (const float*)d_in[44];
    const float* m2_b     = (const float*)d_in[45];

    float* yout = (float*)d_out;

    // ---- workspace layout ----
    float* wsf = (float*)d_ws;
    size_t off = 0;
    auto alloc = [&](size_t n) -> float* {
        float* p = wsf + off;
        off += (n + 63) & ~(size_t)63;
        return p;
    };
    float* h    = alloc((size_t)BB * SEQ * DD);     // LN(x); later reused for xn
    float* swp  = alloc((size_t)BB * SEQ);
    float* cat  = alloc((size_t)BB * MBLK * 3 * DD);
    float* pmap = alloc((size_t)BB * MBLK * DD);
    float* state= alloc(BB * DD);
    float* qc   = alloc(BB * DD);
    float* kc   = alloc((size_t)BB * MBLK * DD);
    float* scs  = alloc(BB * MBLK);
    int*   starts = (int*)alloc(64);
    float* acc  = alloc((size_t)BB * 192 * DD);
    float* ctx  = alloc((size_t)BB * 193 * DD);
    float* nctx = alloc((size_t)BB * 193 * DD);
    float* qkv  = alloc((size_t)BB * 193 * 3 * DD);
    float* mo   = alloc((size_t)BB * 193 * DD);
    float* cls  = alloc(BB * DD);
    float* cn   = alloc(BB * DD);
    float* f1o  = alloc(BB * 4 * DD);
    float* npb  = alloc((size_t)BB * MBLK * DD);
    float* mq   = alloc((size_t)BB * MBLK * DD);
    float* mkv  = alloc((size_t)BB * 192 * 2 * DD);
    float* mo2  = alloc((size_t)BB * MBLK * DD);
    float* dlt  = alloc((size_t)BB * MBLK * DD);
    float* alp  = alloc(64);
    float* sn   = alloc(BB * DD);
    float* orow = alloc(BB * DD);
    const int MCH = 2048;                            // mlp row chunk
    float* m1o  = alloc((size_t)MCH * 4 * DD);

    hipMemsetAsync(acc, 0, (size_t)BB * 192 * DD * sizeof(float), stream);

    auto gemm = [&](const float* A, const float* W, const float* bias, const float* R,
                    float* C, int M, int N, int K, int mode) {
        dim3 g((N + 63) / 64, (M + 63) / 64);
        gemm_wmma_kernel<<<g, 128, 0, stream>>>(A, W, bias, R, C, M, N, K, mode);
    };
    auto ln = [&](const float* X, const float* g, const float* b, const float* extra,
                  int extra_mod, float* Y, int rows) {
        ln_kernel<<<rows, 256, 0, stream>>>(X, g, b, extra, extra_mod, Y);
    };

    // ---------------- trunk ----------------
    // h = LN(x, ln1)
    ln(x, ln1_g, ln1_b, nullptr, 1, h, BB * SEQ);

    // peripheral: scores, stats, sp linear, LN + pos
    matvec_kernel<<<BB * SEQ, 32, 0, stream>>>(h, wp_w, wp_b, swp);
    periph_stats_kernel<<<BB * MBLK, 256, 0, stream>>>(h, swp, cat);
    gemm(cat, sp_w, sp_b, nullptr, pmap, BB * MBLK, DD, 3 * DD, 0);
    ln(pmap, p_ln_g, p_ln_b, pos, MBLK, pmap, BB * MBLK);

    // state = pmap.mean(1)
    mean_kernel<<<(BB * DD + 255) / 256, 256, 0, stream>>>(pmap, state);

    // ---------------- saccade loop ----------------
    for (int t = 0; t < NSAC; ++t) {
        const int Lc = 1 + 64 * (t + 1);
        const int La = 64 * (t + 1);

        // controller q/k + scores + argmax
        gemm(state, cq_w, cq_b, nullptr, qc, BB, DD, DD, 0);
        gemm(pmap, ck_w, ck_b, nullptr, kc, BB * MBLK, DD, DD, 0);
        dot_row_kernel<<<BB * MBLK, 32, 0, stream>>>(qc, kc, scs, 1.0f / 32.0f);
        argmax_kernel<<<BB, 256, 0, stream>>>(scs, starts);

        // append window to acc (acc_new == [acc_old, window])
        gather_window_kernel<<<BB * 64, 256, 0, stream>>>(h, starts, acc, t);

        // ctx = [state, acc_new]
        {
            int total = BB * Lc * DD;
            build_ctx_kernel<<<(total + 255) / 256, 256, 0, stream>>>(state, acc, ctx, Lc);
        }

        // fov self-attention block
        ln(ctx, f_ln1_g, f_ln1_b, nullptr, 1, nctx, BB * Lc);
        gemm(nctx, fa_in_w, fa_in_b, nullptr, qkv, BB * Lc, 3 * DD, DD, 0);
        attn_kernel<<<BB * NHD * Lc, 64, 0, stream>>>(
            qkv, 3 * DD, Lc, qkv + DD, qkv + 2 * DD, 3 * DD, Lc,
            mo, Lc, Lc, 0.125f);
        gemm(mo, fa_out_w, fa_out_b, ctx, ctx, BB * Lc, DD, DD, 2);

        // cls -> FFN -> state
        cls_copy_kernel<<<(BB * DD + 255) / 256, 256, 0, stream>>>(ctx, cls, Lc);
        ln(cls, f_ln2_g, f_ln2_b, nullptr, 1, cn, BB);
        gemm(cn, f1_w, f1_b, nullptr, f1o, BB, 4 * DD, DD, 1);
        gemm(f1o, f2_w, f2_b, cls, state, BB, DD, 4 * DD, 2);

        // map attention: q from LN(pmap), k/v from acc
        ln(pmap, mn_g, mn_b, nullptr, 1, npb, BB * MBLK);
        gemm(npb, ma_in_w, ma_in_b, nullptr, mq, BB * MBLK, DD, DD, 0);
        gemm(acc, ma_in_w + (size_t)DD * DD, ma_in_b + DD, nullptr, mkv,
             BB * 192, 2 * DD, DD, 0);
        attn_kernel<<<BB * NHD * MBLK, 64, 0, stream>>>(
            mq, DD, MBLK, mkv, mkv + DD, 2 * DD, 192,
            mo2, MBLK, La, 0.125f);
        gemm(mo2, ma_out_w, ma_out_b, nullptr, dlt, BB * MBLK, DD, DD, 0);

        // gated update
        alpha_kernel<<<1, 256, 0, stream>>>(g1w, g1b, g2w, g2b, (float)t / (float)NSAC, alp);
        {
            int total = BB * MBLK * DD;
            axpy_kernel<<<(total + 255) / 256, 256, 0, stream>>>(alp, dlt, pmap, total);
        }
    }

    // ---------------- output head ----------------
    // LN(broadcast(state)) == broadcast(LN(state)) row-wise
    ln(state, outn_g, outn_b, nullptr, 1, sn, BB);
    gemm(sn, out_w, out_b, nullptr, orow, BB, DD, DD, 0);
    {
        size_t total = (size_t)BB * SEQ * DD;
        broadcast_add_kernel<<<(unsigned)((total + 255) / 256), 256, 0, stream>>>(x, orow, yout);
    }

    // MLP tail: xn = LN(yout, ln2) stored in h; chunked GEMMs
    ln(yout, ln2_g, ln2_b, nullptr, 1, h, BB * SEQ);
    for (int c = 0; c < (BB * SEQ) / MCH; ++c) {
        const float* xn_c = h + (size_t)c * MCH * DD;
        float* y_c = yout + (size_t)c * MCH * DD;
        gemm(xn_c, m1_w, m1_b, nullptr, m1o, MCH, 4 * DD, DD, 1);
        gemm(m1o, m2_w, m2_b, y_c, y_c, MCH, DD, 4 * DD, 2);
    }
}
